// FeatureDecoder_9216_79568564126068
// MI455X (gfx1250) — matrix operand order repaired and checked by reading, not compiled, not run on hardware
//
#include <hip/hip_runtime.h>
#include <math.h>

// ---------------------------------------------------------------------------
// CDNA5 (gfx1250) wave32 WMMA types
// ---------------------------------------------------------------------------
typedef __attribute__((ext_vector_type(16))) __bf16 v16bf;
typedef __attribute__((ext_vector_type(8)))  __bf16 bf16x8;
typedef __attribute__((ext_vector_type(8)))  float  v8f;

#define WMMA_BF16(a, b, c) \
    __builtin_amdgcn_wmma_f32_16x16x32_bf16(false, (a), false, (b), (short)0, (c), false, false)

// Async global->LDS copy of 16 bytes (per lane), tracked by ASYNCcnt.
// GV addressing: lds-addr VGPR, 64-bit global address VGPR pair.
static __device__ inline void async_copy_b128(unsigned lds_off, const void* gptr) {
    asm volatile("global_load_async_to_lds_b128 %0, %1, off"
                 :: "v"(lds_off), "v"((unsigned long long)(uintptr_t)gptr)
                 : "memory");
}
static __device__ inline void wait_async0() {
    asm volatile("s_wait_asynccnt 0x0" ::: "memory");
}
static __device__ inline unsigned lds_addr(const void* p) {
    return (unsigned)(uintptr_t)p;   // LDS aperture: low 32 bits = wave-relative LDS offset
}

// A-fragment (16x32 bf16, MxK): lane m=lane&15, g=lane>>4 holds K chunks
// [g*8, g*8+8) and [16+g*8, 16+g*8+8)  -> two b128 LDS loads.
static __device__ inline v16bf load_a_frag(const __bf16* lds, int row0, int stride,
                                           int kcol, int lane) {
    const int m = lane & 15, g = lane >> 4;
    const bf16x8 lo = *reinterpret_cast<const bf16x8*>(lds + (row0 + m) * stride + kcol + g * 8);
    const bf16x8 hi = *reinterpret_cast<const bf16x8*>(lds + (row0 + m) * stride + kcol + 16 + g * 8);
    v16bf r;
#pragma unroll
    for (int i = 0; i < 8; ++i) { r[i] = lo[i]; r[i + 8] = hi[i]; }
    return r;
}

// B-fragment (32x16 bf16, KxN) loaded from Bt stored N-major (N x K):
// lane n=lane&15 holds K chunks [g*8, g*8+8) and [16+g*8, 16+g*8+8) -> two b128 LDS loads.
static __device__ inline v16bf load_b_frag(const __bf16* lds, int col0, int stride,
                                           int kcol, int lane) {
    const int n = lane & 15, g = lane >> 4;
    const bf16x8 lo = *reinterpret_cast<const bf16x8*>(lds + (col0 + n) * stride + kcol + g * 8);
    const bf16x8 hi = *reinterpret_cast<const bf16x8*>(lds + (col0 + n) * stride + kcol + 16 + g * 8);
    v16bf r;
#pragma unroll
    for (int i = 0; i < 8; ++i) { r[i] = lo[i]; r[i + 8] = hi[i]; }
    return r;
}

// ---------------------------------------------------------------------------
// Elementwise fp32 -> bf16 convert (8 elems/thread, vectorized)
// ---------------------------------------------------------------------------
__global__ __launch_bounds__(256) void cvt_bf16_kernel(const float* __restrict__ in,
                                                       __bf16* __restrict__ out) {
    const size_t e = ((size_t)blockIdx.x * 256 + threadIdx.x) * 8;
    const float4 f0 = *(const float4*)(in + e);
    const float4 f1 = *(const float4*)(in + e + 4);
    bf16x8 p;
    p[0] = (__bf16)f0.x; p[1] = (__bf16)f0.y; p[2] = (__bf16)f0.z; p[3] = (__bf16)f0.w;
    p[4] = (__bf16)f1.x; p[5] = (__bf16)f1.y; p[6] = (__bf16)f1.z; p[7] = (__bf16)f1.w;
    *reinterpret_cast<bf16x8*>(out + e) = p;
}

// ---------------------------------------------------------------------------
// Weight transpose + fp32 -> bf16 convert: Wt[n*K + k] = W[k*N + n]
// ---------------------------------------------------------------------------
__global__ void transpose_cvt_kernel(const float* __restrict__ W, __bf16* __restrict__ Wt,
                                     int K, int N) {
    int i = blockIdx.x * 256 + threadIdx.x;
    if (i >= K * N) return;
    int k = i / N, n = i - k * N;
    Wt[(size_t)n * K + k] = (__bf16)W[i];
}

// ---------------------------------------------------------------------------
// Tiled WMMA GEMM: out[M x N] = act(A[M x K] @ Bt^T + bias)
//   A:  bf16, row-major M x K
//   Bt: bf16, N x K row-major (pre-transposed weights)
//   8 waves, block tile 128x64, wave tile 32x32 (2x2 WMMA frags), K step 32.
//   Tiles staged via GLOBAL_LOAD_ASYNC_TO_LDS_B128 with double-buffered LDS:
//   per step: s_wait_asynccnt 0 -> barrier -> prefetch next buf -> WMMA cur buf.
// ---------------------------------------------------------------------------
template <bool DO_GELU>
__global__ __launch_bounds__(256) void gemm_kernel(const __bf16* __restrict__ A,
                                                   const __bf16* __restrict__ Bt,
                                                   const float* __restrict__ bias,
                                                   float* __restrict__ out,
                                                   int M, int N, int K) {
    __shared__ __align__(16) __bf16 sA[2][128 * 32];
    __shared__ __align__(16) __bf16 sB[2][64 * 32];

    const int t = threadIdx.x, lane = t & 31, wave = t >> 5;
    const int wm = wave & 3, wn = wave >> 2;           // 4x2 wave grid
    const int m0 = blockIdx.y * 128, n0 = blockIdx.x * 64;

    // per-thread staging assignment (8-elem = 16B granules)
    const int eA0 = t * 8,         rA0 = eA0 >> 5, cA0 = eA0 & 31;   // A granule 0
    const int eA1 = (t + 256) * 8, rA1 = eA1 >> 5, cA1 = eA1 & 31;   // A granule 1
    const int eB  = t * 8,         rB  = eB >> 5,  cB  = eB & 31;    // B granule

    v8f acc[2][2] = {};

    auto prefetch = [&](int buf, int k0) {
        async_copy_b128(lds_addr(&sA[buf][rA0 * 32 + cA0]),
                        A + (size_t)(m0 + rA0) * K + k0 + cA0);
        async_copy_b128(lds_addr(&sA[buf][rA1 * 32 + cA1]),
                        A + (size_t)(m0 + rA1) * K + k0 + cA1);
        async_copy_b128(lds_addr(&sB[buf][rB * 32 + cB]),
                        Bt + (size_t)(n0 + rB) * K + k0 + cB);
    };

    prefetch(0, 0);

    int it = 0;
    for (int k0 = 0; k0 < K; k0 += 32, ++it) {
        const int cur = it & 1;
        wait_async0();          // our async loads into sX[cur] have landed
        __syncthreads();        // everyone's landed; everyone done reading sX[cur^1]
        if (k0 + 32 < K) prefetch(cur ^ 1, k0 + 32);

        v16bf bfr[2];
#pragma unroll
        for (int nt = 0; nt < 2; ++nt)
            bfr[nt] = load_b_frag(sB[cur], wn * 32 + nt * 16, 32, 0, lane);
#pragma unroll
        for (int mt = 0; mt < 2; ++mt) {
            v16bf a = load_a_frag(sA[cur], wm * 32 + mt * 16, 32, 0, lane);
#pragma unroll
            for (int nt = 0; nt < 2; ++nt)
                acc[mt][nt] = WMMA_BF16(a, bfr[nt], acc[mt][nt]);
        }
    }

    // epilogue: bias (+ exact GELU) in fp32.  C/D layout: VGPR r -> M = r + (lane>>4)*8, N = lane&15
#pragma unroll
    for (int mt = 0; mt < 2; ++mt) {
#pragma unroll
        for (int nt = 0; nt < 2; ++nt) {
            const int col = n0 + wn * 32 + nt * 16 + (lane & 15);
            const float bv = bias[col];
#pragma unroll
            for (int r = 0; r < 8; ++r) {
                const int row = m0 + wm * 32 + mt * 16 + r + ((lane >> 4) << 3);
                float v = acc[mt][nt][r] + bv;
                if (DO_GELU) v = 0.5f * v * (1.0f + erff(v * 0.70710678118654752f));
                out[(size_t)row * N + col] = v;
            }
        }
    }
}

// ---------------------------------------------------------------------------
// LayerNorm over last dim (fp32 in, bf16 out). One 256-thread block per row.
// ---------------------------------------------------------------------------
__global__ __launch_bounds__(256) void ln_kernel(const float* __restrict__ in,
                                                 const float* __restrict__ g,
                                                 const float* __restrict__ be,
                                                 __bf16* __restrict__ out, int N) {
    const int row = blockIdx.x, t = threadIdx.x;
    __shared__ float s1[256], s2[256];
    float vloc[3];
    float s = 0.f, ss = 0.f;
    int cnt = 0;
    for (int i = t; i < N; i += 256) {
        float v = in[(size_t)row * N + i];
        vloc[cnt++] = v;
        s += v; ss += v * v;
    }
    s1[t] = s; s2[t] = ss;
    __syncthreads();
    for (int o = 128; o > 0; o >>= 1) {
        if (t < o) { s1[t] += s1[t + o]; s2[t] += s2[t + o]; }
        __syncthreads();
    }
    const float mean = s1[0] / (float)N;
    const float var  = s2[0] / (float)N - mean * mean;
    const float inv  = rsqrtf(var + 1e-5f);
    cnt = 0;
    for (int i = t; i < N; i += 256)
        out[(size_t)row * N + i] = (__bf16)((vloc[cnt++] - mean) * inv * g[i] + be[i]);
}

// ---------------------------------------------------------------------------
// SE channel attention helpers
// ---------------------------------------------------------------------------
__global__ void zero_kernel(float* p, int n) {
    int i = blockIdx.x * 256 + threadIdx.x;
    if (i < n) p[i] = 0.f;
}

// pooled[b][c] += sum over a 256-row chunk of y[b][n][c].  grid = (9*36, 8), block = 128.
__global__ __launch_bounds__(128) void pool_kernel(const float* __restrict__ y,
                                                   float* __restrict__ pooled) {
    const int b = blockIdx.y;
    const int cblk = blockIdx.x % 9, nch = blockIdx.x / 9;
    const int c = cblk * 128 + threadIdx.x;
    const float* p = y + ((size_t)b * 9216 + (size_t)nch * 256) * 1152 + c;
    float s = 0.f;
    for (int n = 0; n < 256; ++n) s += p[(size_t)n * 1152];
    atomicAdd(&pooled[b * 1152 + c], s);
}

// s[b][c] = sigmoid(relu(mean_pool @ w1 + b1) @ w2 + b2).  One block per batch.
__global__ __launch_bounds__(256) void se_kernel(const float* __restrict__ pooled,
                                                 const float* __restrict__ w1,
                                                 const float* __restrict__ b1,
                                                 const float* __restrict__ w2,
                                                 const float* __restrict__ b2,
                                                 float* __restrict__ svec) {
    const int b = blockIdx.x, t = threadIdx.x;
    __shared__ float sp[1152];
    __shared__ float st1[144];
    for (int i = t; i < 1152; i += 256) sp[i] = pooled[b * 1152 + i] * (1.0f / 9216.0f);
    __syncthreads();
    if (t < 144) {
        float a = b1[t];
        for (int k = 0; k < 1152; ++k) a += sp[k] * w1[k * 144 + t];
        st1[t] = a > 0.f ? a : 0.f;
    }
    __syncthreads();
    for (int c = t; c < 1152; c += 256) {
        float a = b2[c];
        for (int k = 0; k < 144; ++k) a += st1[k] * w2[k * 1152 + c];
        svec[b * 1152 + c] = 1.0f / (1.0f + expf(-a));
    }
}

// ---------------------------------------------------------------------------
// Windowed self-attention (window=32, scale 1/sqrt(1152)), SE scale fused.
// One 256-thread block (8 waves) per window.  Scores via WMMA over 9 channel
// chunks of 128; fp32 softmax; attn @ yw via WMMA on a transposed LDS chunk.
// ---------------------------------------------------------------------------
__global__ __launch_bounds__(256) void attn_kernel(const float* __restrict__ y,
                                                   const float* __restrict__ svec,
                                                   float* __restrict__ out) {
    const int blk = blockIdx.x;
    const int b = blk / 288, win = blk % 288;
    const int t = threadIdx.x, lane = t & 31, wave = t >> 5;
    const float* yw = y + ((size_t)b * 9216 + (size_t)win * 32) * 1152;
    const float* sv = svec + b * 1152;

    __shared__ __align__(16) __bf16 sX[32 * 128];   // row-major chunk, reused transposed
    __shared__ float  sS[32 * 32];
    __shared__ __align__(16) __bf16 sAttn[32 * 32];

    v8f acc[2][2] = {};

    // ---- Phase 1: scores = (s*yw) @ (s*yw)^T, accumulated over channel chunks
    for (int cc = 0; cc < 9; ++cc) {
#pragma unroll
        for (int i = 0; i < 2; ++i) {
            const int v = t + i * 256, e = v * 8;
            const int r = e >> 7, c = e & 127, ch = cc * 128 + c;
            const float4 f0 = *(const float4*)(yw + (size_t)r * 1152 + ch);
            const float4 f1 = *(const float4*)(yw + (size_t)r * 1152 + ch + 4);
            bf16x8 p;
            p[0] = (__bf16)(f0.x * sv[ch + 0]); p[1] = (__bf16)(f0.y * sv[ch + 1]);
            p[2] = (__bf16)(f0.z * sv[ch + 2]); p[3] = (__bf16)(f0.w * sv[ch + 3]);
            p[4] = (__bf16)(f1.x * sv[ch + 4]); p[5] = (__bf16)(f1.y * sv[ch + 5]);
            p[6] = (__bf16)(f1.z * sv[ch + 6]); p[7] = (__bf16)(f1.w * sv[ch + 7]);
            *reinterpret_cast<bf16x8*>(&sX[r * 128 + c]) = p;
        }
        __syncthreads();
        if (wave == 0) {
#pragma unroll
            for (int kk = 0; kk < 4; ++kk) {
                v16bf a0 = load_a_frag(sX, 0, 128, kk * 32, lane);
                v16bf a1 = load_a_frag(sX, 16, 128, kk * 32, lane);
                v16bf b0 = load_b_frag(sX, 0, 128, kk * 32, lane);
                v16bf b1 = load_b_frag(sX, 16, 128, kk * 32, lane);
                acc[0][0] = WMMA_BF16(a0, b0, acc[0][0]);
                acc[0][1] = WMMA_BF16(a0, b1, acc[0][1]);
                acc[1][0] = WMMA_BF16(a1, b0, acc[1][0]);
                acc[1][1] = WMMA_BF16(a1, b1, acc[1][1]);
            }
        }
        __syncthreads();
    }

    if (wave == 0) {
        const float scale = 0.02946278254943948f;   // 1/sqrt(1152)
#pragma unroll
        for (int mt = 0; mt < 2; ++mt)
#pragma unroll
            for (int nt = 0; nt < 2; ++nt)
#pragma unroll
                for (int r = 0; r < 8; ++r) {
                    const int q = mt * 16 + r + ((lane >> 4) << 3);
                    const int k = nt * 16 + (lane & 15);
                    sS[q * 32 + k] = acc[mt][nt][r] * scale;
                }
    }
    __syncthreads();

    // ---- softmax (fp32) over keys, emit bf16 attn weights
    if (t < 32) {
        float mx = -1e30f;
        for (int j = 0; j < 32; ++j) mx = fmaxf(mx, sS[t * 32 + j]);
        float sum = 0.f;
        for (int j = 0; j < 32; ++j) sum += expf(sS[t * 32 + j] - mx);
        const float inv = 1.0f / sum;
        for (int j = 0; j < 32; ++j)
            sAttn[t * 32 + j] = (__bf16)(expf(sS[t * 32 + j] - mx) * inv);
    }
    __syncthreads();

    // ---- Phase 2: out = attn @ (s*yw); transposed chunk so B-frags are contiguous
    float* outw = out + ((size_t)b * 9216 + (size_t)win * 32) * 1152;
    for (int cc = 0; cc < 9; ++cc) {
#pragma unroll
        for (int i = 0; i < 2; ++i) {
            const int v = t + i * 256, e = v * 8;
            const int k = e >> 7, c = e & 127, ch = cc * 128 + c;
            const float4 f0 = *(const float4*)(yw + (size_t)k * 1152 + ch);
            const float4 f1 = *(const float4*)(yw + (size_t)k * 1152 + ch + 4);
            float vv[8] = {f0.x, f0.y, f0.z, f0.w, f1.x, f1.y, f1.z, f1.w};
#pragma unroll
            for (int j = 0; j < 8; ++j)
                sX[(c + j) * 32 + k] = (__bf16)(vv[j] * sv[ch + j]);
        }
        __syncthreads();
#pragma unroll
        for (int mt = 0; mt < 2; ++mt) {
            v16bf a  = load_a_frag(sAttn, mt * 16, 32, 0, lane);
            v16bf bf = load_b_frag(sX, wave * 16, 32, 0, lane);
            v8f z = {};
            v8f o = WMMA_BF16(a, bf, z);
#pragma unroll
            for (int r = 0; r < 8; ++r) {
                const int tok = mt * 16 + r + ((lane >> 4) << 3);
                const int ch  = cc * 128 + wave * 16 + (lane & 15);
                outw[(size_t)tok * 1152 + ch] = o[r];
            }
        }
        __syncthreads();
    }
}

// ---------------------------------------------------------------------------
// Host-side orchestration
// ---------------------------------------------------------------------------
extern "C" void kernel_launch(void* const* d_in, const int* in_sizes, int n_in,
                              void* d_out, int out_size, void* d_ws, size_t ws_size,
                              hipStream_t stream) {
    (void)in_sizes; (void)n_in; (void)out_size; (void)ws_size;

    constexpr int B = 8, NSEQ = 9216, C = 1152;
    constexpr int H1 = 768, H2 = 512, H3 = 256;
    constexpr int MT = B * NSEQ;                 // 73728 tokens

    const float* x     = (const float*)d_in[0];
    const float* W1    = (const float*)d_in[1];
    const float* b1    = (const float*)d_in[2];
    const float* g1    = (const float*)d_in[3];
    const float* be1   = (const float*)d_in[4];
    const float* W2    = (const float*)d_in[5];
    const float* b2    = (const float*)d_in[6];
    const float* g2    = (const float*)d_in[7];
    const float* be2   = (const float*)d_in[8];
    const float* W3    = (const float*)d_in[9];
    const float* b3    = (const float*)d_in[10];
    const float* g3    = (const float*)d_in[11];
    const float* be3   = (const float*)d_in[12];
    const float* Wo    = (const float*)d_in[13];
    const float* bo    = (const float*)d_in[14];
    const float* ca_w1 = (const float*)d_in[15];
    const float* ca_b1 = (const float*)d_in[16];
    const float* ca_w2 = (const float*)d_in[17];
    const float* ca_b2 = (const float*)d_in[18];
    float* outp = (float*)d_out;

    char* ws = (char*)d_ws;
    size_t off = 0;
    auto alloc = [&](size_t bytes) -> void* {
        void* p = ws + off;
        off += (bytes + 255) & ~(size_t)255;
        return p;
    };

    __bf16* xbf = (__bf16*)alloc((size_t)MT * C * 2);
    __bf16* Wt1 = (__bf16*)alloc((size_t)C * H1 * 2);
    __bf16* Wt2 = (__bf16*)alloc((size_t)H1 * H2 * 2);
    __bf16* Wt3 = (__bf16*)alloc((size_t)H2 * H3 * 2);
    __bf16* Wto = (__bf16*)alloc((size_t)H3 * C * 2);
    float*  hpre   = (float*)alloc((size_t)MT * H1 * 4);   // reused for all pre-LN acts
    __bf16* hbf    = (__bf16*)alloc((size_t)MT * H1 * 2);  // reused for all LN'd acts
    float*  yb     = (float*)alloc((size_t)MT * C * 4);
    float*  pooled = (float*)alloc((size_t)B * C * 4);
    float*  svec   = (float*)alloc((size_t)B * C * 4);

    // x -> bf16 (one streaming pass), weight transpose + bf16 convert
    cvt_bf16_kernel<<<(int)(((size_t)MT * C / 8) / 256), 256, 0, stream>>>(x, xbf);
    transpose_cvt_kernel<<<(C * H1 + 255) / 256, 256, 0, stream>>>(W1, Wt1, C, H1);
    transpose_cvt_kernel<<<(H1 * H2 + 255) / 256, 256, 0, stream>>>(W2, Wt2, H1, H2);
    transpose_cvt_kernel<<<(H2 * H3 + 255) / 256, 256, 0, stream>>>(W3, Wt3, H2, H3);
    transpose_cvt_kernel<<<(H3 * C + 255) / 256, 256, 0, stream>>>(Wo, Wto, H3, C);

    // block 1: x @ W1 -> GELU -> LN
    gemm_kernel<true><<<dim3(H1 / 64, MT / 128), 256, 0, stream>>>(xbf, Wt1, b1, hpre, MT, H1, C);
    ln_kernel<<<MT, 256, 0, stream>>>(hpre, g1, be1, hbf, H1);
    // block 2
    gemm_kernel<true><<<dim3(H2 / 64, MT / 128), 256, 0, stream>>>(hbf, Wt2, b2, hpre, MT, H2, H1);
    ln_kernel<<<MT, 256, 0, stream>>>(hpre, g2, be2, hbf, H2);
    // block 3
    gemm_kernel<true><<<dim3(H3 / 64, MT / 128), 256, 0, stream>>>(hbf, Wt3, b3, hpre, MT, H3, H2);
    ln_kernel<<<MT, 256, 0, stream>>>(hpre, g3, be3, hbf, H3);
    // output projection (no GELU)
    gemm_kernel<false><<<dim3(C / 64, MT / 128), 256, 0, stream>>>(hbf, Wto, bo, yb, MT, C, H3);

    // SE channel attention
    zero_kernel<<<(B * C + 255) / 256, 256, 0, stream>>>(pooled, B * C);
    pool_kernel<<<dim3(9 * 36, B), 128, 0, stream>>>(yb, pooled);
    se_kernel<<<B, 256, 0, stream>>>(pooled, ca_w1, ca_b1, ca_w2, ca_b2, svec);

    // windowed self-attention, SE scale fused
    attn_kernel<<<B * (NSEQ / 32), 256, 0, stream>>>(yb, svec, outp);
}
